// SEBlock_6004364280589
// MI455X (gfx1250) — matrix operand order repaired and checked by reading, not compiled, not run on hardware
//
#include <hip/hip_runtime.h>
#include <hip/hip_bf16.h>
#include <math.h>

typedef __attribute__((ext_vector_type(16))) _Float16 v16h;
typedef __attribute__((ext_vector_type(8)))  float    v8f;
typedef __attribute__((ext_vector_type(4)))  float    v4f;

#define IN_CH 256
#define SQ_CH 64
#define HW    12544   // 112*112
#define HW4   3136    // HW/4
#define NB    32

// ---------------- Kernel 1: global average pool -> s[n*IN_CH + c] -------------
__global__ void se_pool_kernel(const float* __restrict__ x, float* __restrict__ s) {
    __shared__ float red[256];
    const int b = blockIdx.x;                       // flat (n*IN_CH + c)
    const v4f* xb = (const v4f*)(x + (size_t)b * HW);
    float sum = 0.f;
    for (int i = threadIdx.x; i < HW4; i += 256) {
        v4f v = __builtin_nontemporal_load(&xb[i]); // streaming: no reuse, keep L2 clean
        sum += v.x + v.y + v.z + v.w;
    }
    red[threadIdx.x] = sum;
    __syncthreads();
    for (int off = 128; off > 0; off >>= 1) {
        if (threadIdx.x < off) red[threadIdx.x] += red[threadIdx.x + off];
        __syncthreads();
    }
    if (threadIdx.x == 0) s[b] = red[0] * (1.0f / (float)HW);
}

// ---------------- Kernel 2: SE MLP via WMMA (1 block, 8 waves) ----------------
// r = relu(S x WrT + br) ; gate = sigmoid(r x WeT + be)
__global__ void se_mlp_kernel(const float* __restrict__ s,
                              const float* __restrict__ w_reduce,   // [64][256]
                              const float* __restrict__ b_reduce,   // [64]
                              const float* __restrict__ w_expand,   // [256][64]
                              const float* __restrict__ b_expand,   // [256]
                              float* __restrict__ gate) {           // [32][256]
    __shared__ _Float16 sR[NB * SQ_CH];             // 32x64 relu output, 4 KB
    const int lane = threadIdx.x & 31;
    const int wave = threadIdx.x >> 5;
    const int hi   = (lane >= 16) ? 1 : 0;          // upper half-wave
    const int lcol = lane & 15;

    // ---- GEMM1: 32x64 = [32x256] x [256x64]; one 16x16 tile per wave --------
    {
        const int m0 = (wave >> 2) * 16;
        const int n0 = (wave & 3) * 16;
        v8f acc = {};
        for (int kk = 0; kk < IN_CH; kk += 32) {
            v16h a, bm;
            // A fragment: lanes<16 hold K=0..15, >=16 K=16..31
            const int m  = m0 + lcol;
            const int ka = kk + (hi ? 16 : 0);
            #pragma unroll
            for (int j = 0; j < 8; ++j) {
                a[j]     = (_Float16)s[m * IN_CH + ka + j];
                a[8 + j] = (_Float16)s[m * IN_CH + ka + 8 + j];
            }
            // B fragment: B(k,n) = w_reduce[n][k]; lanes<16 hold K=0..15, >=16 K=16..31
            const int kb = kk + (hi ? 16 : 0);
            const float* wr = w_reduce + (size_t)(n0 + lcol) * IN_CH;
            #pragma unroll
            for (int j = 0; j < 16; ++j) bm[j] = (_Float16)wr[kb + j];
            acc = __builtin_amdgcn_wmma_f32_16x16x32_f16(false, a, false, bm,
                                                         (short)0, acc, false, false);
        }
        // bias + relu -> LDS (C/D layout: M = r + 8*hi, N = lcol)
        const int n = n0 + lcol;
        const float bias = b_reduce[n];
        #pragma unroll
        for (int r = 0; r < 8; ++r) {
            const int m = m0 + r + hi * 8;
            float v = acc[r] + bias;
            sR[m * SQ_CH + n] = (_Float16)(v > 0.f ? v : 0.f);
        }
    }
    __syncthreads();

    // ---- GEMM2: 32x256 = [32x64] x [64x256]; 32 tiles, 4 per wave -----------
    for (int t = wave * 4; t < wave * 4 + 4; ++t) {
        const int m0 = (t >> 4) * 16;
        const int n0 = (t & 15) * 16;
        v8f acc = {};
        for (int kk = 0; kk < SQ_CH; kk += 32) {
            v16h a, bm;
            const int m  = m0 + lcol;
            const int ka = kk + (hi ? 16 : 0);
            #pragma unroll
            for (int j = 0; j < 8; ++j) {
                a[j]     = sR[m * SQ_CH + ka + j];
                a[8 + j] = sR[m * SQ_CH + ka + 8 + j];
            }
            const int kb = kk + (hi ? 16 : 0);
            const float* we = w_expand + (size_t)(n0 + lcol) * SQ_CH;
            #pragma unroll
            for (int j = 0; j < 16; ++j) bm[j] = (_Float16)we[kb + j];
            acc = __builtin_amdgcn_wmma_f32_16x16x32_f16(false, a, false, bm,
                                                         (short)0, acc, false, false);
        }
        const int n = n0 + lcol;
        const float bias = b_expand[n];
        #pragma unroll
        for (int r = 0; r < 8; ++r) {
            const int m = m0 + r + hi * 8;
            float g = acc[r] + bias;
            gate[m * IN_CH + n] = 1.0f / (1.0f + __expf(-g));
        }
    }
}

// ---------------- Kernel 3: out = gate * x (streaming) ------------------------
__global__ void se_scale_kernel(const float* __restrict__ x,
                                const float* __restrict__ gate,
                                float* __restrict__ out) {
    const int b = blockIdx.x;                       // flat (n*IN_CH + c)
    const float g = gate[b];
    const v4f* xb = (const v4f*)(x + (size_t)b * HW);
    v4f* ob = (v4f*)(out + (size_t)b * HW);
    for (int i = threadIdx.x; i < HW4; i += 256) {
        v4f v = __builtin_nontemporal_load(&xb[i]);
        v *= g;
        __builtin_nontemporal_store(v, &ob[i]);
    }
}

extern "C" void kernel_launch(void* const* d_in, const int* in_sizes, int n_in,
                              void* d_out, int out_size, void* d_ws, size_t ws_size,
                              hipStream_t stream) {
    const float* x        = (const float*)d_in[0];
    const float* w_reduce = (const float*)d_in[1];
    const float* b_reduce = (const float*)d_in[2];
    const float* w_expand = (const float*)d_in[3];
    const float* b_expand = (const float*)d_in[4];
    float* out = (float*)d_out;

    float* s    = (float*)d_ws;          // [32*256] fp32
    float* gate = s + NB * IN_CH;        // [32*256] fp32

    const int nc = in_sizes[0] / HW;     // = 32*256 = 8192 channel planes

    se_pool_kernel <<<nc, 256, 0, stream>>>(x, s);
    se_mlp_kernel  <<<1, 256, 0, stream>>>(s, w_reduce, b_reduce, w_expand, b_expand, gate);
    se_scale_kernel<<<nc, 256, 0, stream>>>(x, gate, out);
}
